// MetaSlot_20890720928570
// MI455X (gfx1250) — matrix operand order repaired and checked by reading, not compiled, not run on hardware
//
#include <hip/hip_runtime.h>
#include <hip/hip_bf16.h>
#include <stdint.h>

typedef __bf16 bf16_t;
typedef __attribute__((ext_vector_type(16))) __bf16 v16bf;
typedef __attribute__((ext_vector_type(8)))  __bf16 v8bf;
typedef __attribute__((ext_vector_type(8)))  float  v8f;

#define EMBED   512
#define KSTEPS  (EMBED / 32)    // 16 k-steps of 32 for bf16 WMMA
#define BROWPAD 520             // 512 + 8 elem pad -> 1040B row stride (bank stagger)
#define MATSZ   (16 * BROWPAD)  // elements per (buf, hi/lo) LDS region
#define LDS_BYTES (4 * MATSZ * (int)sizeof(bf16_t))   // 2 bufs x (hi,lo)

// ---------------------------------------------------------------------------
// Split f32 -> (bf16 hi, bf16 lo) so that hi + lo ~= x with ~2^-18 rel error.
// ---------------------------------------------------------------------------
__global__ __launch_bounds__(256) void split_hi_lo(const float* __restrict__ x,
                                                   bf16_t* __restrict__ hi,
                                                   bf16_t* __restrict__ lo,
                                                   int n) {
    int i = blockIdx.x * blockDim.x + threadIdx.x;
    if (i < n) {
        float v = x[i];
        bf16_t h = (bf16_t)v;               // RNE to bf16
        hi[i] = h;
        lo[i] = (bf16_t)(v - (float)h);     // residual
    }
}

// ---------------------------------------------------------------------------
// ||t||^2 per template row, exact f32. One wave per row.
// ---------------------------------------------------------------------------
__global__ __launch_bounds__(32) void template_sq(const float* __restrict__ templat,
                                                  float* __restrict__ tsq) {
    int row  = blockIdx.x;
    int lane = threadIdx.x;
    const float* p = templat + (size_t)row * EMBED;
    float s = 0.0f;
    for (int j = lane; j < EMBED; j += 32) {
        float v = p[j];
        s += v * v;
    }
    #pragma unroll
    for (int off = 16; off > 0; off >>= 1)
        s += __shfl_xor(s, off, 32);
    if (lane == 0) tsq[row] = s;
}

// ---------------------------------------------------------------------------
// Fused GEMM (bf16x3 emulated-f32) + per-row argmax.
// Block = 256 threads = 8 waves; each wave owns 16 rows; grid.x = n_tok/128.
// A panel (16 x 512, hi+lo) resident in VGPRs.
// B tiles double-buffered in LDS via async global->LDS loads (ASYNCcnt path),
// shared by all 8 waves: L2 read traffic drops 8x vs per-wave streaming.
// Three independent accumulators (hi.hi / hi.lo / lo.hi) break the WMMA->WMMA
// RAW chain so the XDL pipe can issue back-to-back without delay-alu stalls.
// ---------------------------------------------------------------------------
__global__ __launch_bounds__(256) void vq_gemm_argmax(
    const bf16_t* __restrict__ enc_hi, const bf16_t* __restrict__ enc_lo,
    const bf16_t* __restrict__ tem_hi, const bf16_t* __restrict__ tem_lo,
    const float*  __restrict__ tsq,    int n_emb,
    int* __restrict__ zidx) {

    extern __shared__ bf16_t smem[];   // [buf*2 + mat][row*BROWPAD + k]

    const int tid  = threadIdx.x;
    const int lane = tid & 31;
    const int wave = tid >> 5;
    const int half = lane >> 4;          // 0: lanes 0-15, 1: lanes 16-31
    const int l15  = lane & 15;
    const int rowBase = blockIdx.x * 128 + wave * 16;

    // ---- Loader decomposition: 256 threads x one 16B chunk per tile ----
    const int mat = tid >> 7;            // 0 = hi matrix, 1 = lo matrix
    const int rr  = (tid >> 3) & 15;     // row within 16-col tile
    const int cc  = tid & 7;             // 16-byte chunk within 1KB row
    const bf16_t* gsrc_base =
        (mat ? tem_lo : tem_hi) + (size_t)rr * EMBED + cc * 8;
    // LDS byte offsets (addr[31:0] of the generic pointer is the LDS offset)
    uint32_t lds_dst[2];
    lds_dst[0] = (uint32_t)(uintptr_t)&smem[(0 * 2 + mat) * MATSZ + rr * BROWPAD + cc * 8];
    lds_dst[1] = (uint32_t)(uintptr_t)&smem[(1 * 2 + mat) * MATSZ + rr * BROWPAD + cc * 8];

    // ---- Load resident A panel (contiguous k order: element s <-> k = 16*half + s) ----
    v16bf a_hi[KSTEPS], a_lo[KSTEPS];
    {
        const size_t arow = (size_t)(rowBase + l15) * EMBED;
        #pragma unroll
        for (int ks = 0; ks < KSTEPS; ++ks) {
            const int kb = ks * 32 + half * 16;
            v8bf h0 = *(const v8bf*)(enc_hi + arow + kb);
            v8bf h1 = *(const v8bf*)(enc_hi + arow + kb + 8);
            v8bf l0 = *(const v8bf*)(enc_lo + arow + kb);
            v8bf l1 = *(const v8bf*)(enc_lo + arow + kb + 8);
            #pragma unroll
            for (int j = 0; j < 8; ++j) {
                a_hi[ks][j]     = h0[j];
                a_hi[ks][8 + j] = h1[j];
                a_lo[ks][j]     = l0[j];
                a_lo[ks][8 + j] = l1[j];
            }
        }
    }

    // Running argmax in C layout: acc[r] <-> (row = r + 8*half, col = tile*16 + l15)
    float bestv[8];
    int   besti[8];
    #pragma unroll
    for (int r = 0; r < 8; ++r) { bestv[r] = -3.4e38f; besti[r] = 0; }

    const int ntiles = n_emb >> 4;

    // ---- Prologue: async-load tile 0 into buffer 0 ----
    {
        const bf16_t* g = gsrc_base;     // tile 0
        asm volatile("global_load_async_to_lds_b128 %0, %1, off"
                     :: "v"(lds_dst[0]), "v"(g) : "memory");
    }
    asm volatile("s_wait_asynccnt 0x0" ::: "memory");
    __syncthreads();

    for (int t = 0; t < ntiles; ++t) {
        const int cur = t & 1;
        const int nxt = cur ^ 1;

        // Prefetch next B tile into the other buffer via the async engine.
        if (t + 1 < ntiles) {
            const bf16_t* g = gsrc_base + (size_t)(t + 1) * (16 * EMBED);
            asm volatile("global_load_async_to_lds_b128 %0, %1, off"
                         :: "v"(lds_dst[nxt]), "v"(g) : "memory");
        }

        // ---- Compute tile t from LDS ----
        const bf16_t* bh = &smem[(cur * 2 + 0) * MATSZ + l15 * BROWPAD + half * 16];
        const bf16_t* bl = &smem[(cur * 2 + 1) * MATSZ + l15 * BROWPAD + half * 16];
        v8f acc0 = {0.f, 0.f, 0.f, 0.f, 0.f, 0.f, 0.f, 0.f};
        v8f acc1 = acc0;
        v8f acc2 = acc0;
        #pragma unroll
        for (int ks = 0; ks < KSTEPS; ++ks) {
            v8bf bh0 = *(const v8bf*)(bh + ks * 32);
            v8bf bh1 = *(const v8bf*)(bh + ks * 32 + 8);
            v8bf bl0 = *(const v8bf*)(bl + ks * 32);
            v8bf bl1 = *(const v8bf*)(bl + ks * 32 + 8);
            v16bf Bh, Bl;
            #pragma unroll
            for (int j = 0; j < 8; ++j) {
                Bh[j] = bh0[j]; Bh[8 + j] = bh1[j];
                Bl[j] = bl0[j]; Bl[8 + j] = bl1[j];
            }
            // e.t ~= e_hi.t_hi + e_hi.t_lo + e_lo.t_hi  (drop lo.lo ~ 2^-18)
            // Independent accumulators -> no WMMA->WMMA RAW in program order.
            acc0 = __builtin_amdgcn_wmma_f32_16x16x32_bf16(false, a_hi[ks], false, Bh,
                                                           (short)0, acc0, false, false);
            acc1 = __builtin_amdgcn_wmma_f32_16x16x32_bf16(false, a_hi[ks], false, Bl,
                                                           (short)0, acc1, false, false);
            acc2 = __builtin_amdgcn_wmma_f32_16x16x32_bf16(false, a_lo[ks], false, Bh,
                                                           (short)0, acc2, false, false);
        }

        const int   col = t * 16 + l15;
        const float ts  = tsq[col];
        #pragma unroll
        for (int r = 0; r < 8; ++r) {
            float s = 2.0f * (acc0[r] + acc1[r] + acc2[r]) - ts;  // = -(dist) + const(row)
            if (s > bestv[r]) { bestv[r] = s; besti[r] = col; }
        }

        // Publish tile t+1 (this wave's async loads) and protect buffer reuse.
        asm volatile("s_wait_asynccnt 0x0" ::: "memory");
        __syncthreads();
    }

    // ---- Butterfly argmax across the 16 lanes sharing a row ----
    #pragma unroll
    for (int r = 0; r < 8; ++r) {
        #pragma unroll
        for (int off = 1; off <= 8; off <<= 1) {
            float ov = __shfl_xor(bestv[r], off, 32);
            int   oi = __shfl_xor(besti[r], off, 32);
            if (ov > bestv[r] || (ov == bestv[r] && oi < besti[r])) {
                bestv[r] = ov;
                besti[r] = oi;
            }
        }
        if (l15 == 0)
            zidx[rowBase + half * 8 + r] = besti[r];
    }
}

// ---------------------------------------------------------------------------
// quant[n] = templat[zidx[n]] — 2 rows per 256-thread block, float4 copies.
// ---------------------------------------------------------------------------
__global__ __launch_bounds__(256) void vq_gather(const float* __restrict__ templat,
                                                 const int* __restrict__ zidx,
                                                 float* __restrict__ quant) {
    int row = blockIdx.x * 2 + (threadIdx.x >> 7);
    int t   = threadIdx.x & 127;
    int idx = zidx[row];
    const float4* src = (const float4*)(templat + (size_t)idx * EMBED);
    float4*       dst = (float4*)(quant + (size_t)row * EMBED);
    dst[t] = src[t];
}

// ---------------------------------------------------------------------------
static inline size_t align256(size_t x) { return (x + 255) & ~(size_t)255; }

extern "C" void kernel_launch(void* const* d_in, const int* in_sizes, int n_in,
                              void* d_out, int out_size, void* d_ws, size_t ws_size,
                              hipStream_t stream) {
    const float* encode  = (const float*)d_in[0];
    const float* templat = (const float*)d_in[1];
    const int n_tok = in_sizes[0] / EMBED;   // 16384
    const int n_emb = in_sizes[1] / EMBED;   // 2048

    float* quant = (float*)d_out;
    int*   zidx  = (int*)((float*)d_out + (size_t)n_tok * EMBED);

    // Workspace layout
    const size_t enc_elems = (size_t)n_tok * EMBED;
    const size_t tem_elems = (size_t)n_emb * EMBED;
    char* ws = (char*)d_ws;
    bf16_t* enc_hi = (bf16_t*)ws;  ws += align256(enc_elems * sizeof(bf16_t));
    bf16_t* enc_lo = (bf16_t*)ws;  ws += align256(enc_elems * sizeof(bf16_t));
    bf16_t* tem_hi = (bf16_t*)ws;  ws += align256(tem_elems * sizeof(bf16_t));
    bf16_t* tem_lo = (bf16_t*)ws;  ws += align256(tem_elems * sizeof(bf16_t));
    float*  tsq    = (float*)ws;   ws += align256((size_t)n_emb * sizeof(float));

    split_hi_lo<<<(int)((enc_elems + 255) / 256), 256, 0, stream>>>(encode, enc_hi, enc_lo, (int)enc_elems);
    split_hi_lo<<<(int)((tem_elems + 255) / 256), 256, 0, stream>>>(templat, tem_hi, tem_lo, (int)tem_elems);
    template_sq<<<n_emb, 32, 0, stream>>>(templat, tsq);

    vq_gemm_argmax<<<n_tok / 128, 256, LDS_BYTES, stream>>>(enc_hi, enc_lo, tem_hi, tem_lo,
                                                            tsq, n_emb, zidx);

    vq_gather<<<n_tok / 2, 256, 0, stream>>>(templat, zidx, quant);
}
